// SAGE_78469052498036
// MI455X (gfx1250) — matrix operand order repaired and checked by reading, not compiled, not run on hardware
//
#include <hip/hip_runtime.h>
#include <math.h>

typedef __attribute__((ext_vector_type(16))) __bf16 v16bf;
typedef __attribute__((ext_vector_type(8)))  float  v8f;

#define NN    32768
#define EE    65536
#define NDSTN 4096
#define DINN  128
#define DHH   256
#define KKK   4096
#define DOUTN 40

// ---------------- workspace layout (float offsets) ----------------
static const long long WS_RSQO   = 0;
static const long long WS_RSQI   = WS_RSQO + NN;
static const long long WS_HS     = WS_RSQI + NN;
static const long long WS_MSG    = WS_HS   + (long long)NN * DHH;
static const long long WS_HA     = WS_MSG  + (long long)NN * DHH;
static const long long WS_HB     = WS_HA   + (long long)NN * DHH;
static const long long WS_HCAT   = WS_HB   + (long long)NN * DHH;
static const long long WS_ROWMAX = WS_HCAT + 2LL * NDSTN * DHH;
static const long long WS_ROWINV = WS_ROWMAX + 2LL * NDSTN;
static const long long WS_COLSUM = WS_ROWINV + 2LL * NDSTN;
static const long long WS_QN     = WS_COLSUM + KKK;
static const long long WS_QE     = WS_QN + (long long)NDSTN * DINN;
static const long long WS_ADJQ   = WS_QE + (long long)NDSTN * DHH;
static const long long WS_BNS    = WS_ADJQ + (long long)NDSTN * NDSTN;
static const long long WS_BNSQ   = WS_BNS + DHH;
static const long long WS_SCAL   = WS_BNSQ + DHH;   // [0]=commit [1]=feat [2]=edge ; u[3]=min u[4]=max

// ---------------- output layout (float offsets) ----------------
static const long long O_HOUT  = 0;
static const long long O_LOSS  = (long long)NDSTN * DOUTN;              // 163840
static const long long O_DIST  = O_LOSS + 1;                            // 163841
static const long long O_PERP  = O_DIST + 2LL * NDSTN * KKK;            // +33554432
static const long long O_QUANT = O_PERP + 1;

// ---------------- helpers ----------------
__device__ inline unsigned f2ord(float f) {
  unsigned u = __float_as_uint(f);
  return (u & 0x80000000u) ? ~u : (u | 0x80000000u);
}
__device__ inline float ord2f(unsigned u) {
  return (u & 0x80000000u) ? __uint_as_float(u & 0x7fffffffu) : __uint_as_float(~u);
}

union FragU { unsigned u[8]; v16bf v; };

// =====================================================================
//  WMMA bf16 GEMM:  C = post( A @ B + bias )
//  FLAGS (compile-time): 1=relu  2=negate output
//                        4=A := softmax(-A) via rowmax/rowinv
//                        8=B transposed (B[k][n] = Bsrc[n*ldb+k])
//                        16=A := -A
//  ALIGNED (compile-time): M%64==0 && Nc%128==0 && Kd%32==0 -> no guards,
//  straight-line staging (no per-element exec masking).
//  block = 256 threads (8 waves), tile 64(M) x 128(N) x 32(K)
//  B tile kept TRANSPOSED in LDS so each fragment VGPR is one 32-bit load.
// =====================================================================
template<int FLAGS, bool ALIGNED>
__global__ __launch_bounds__(256)
void k_wmma_gemm(const float* __restrict__ A, const float* __restrict__ B,
                 const float* __restrict__ bias, float* __restrict__ C,
                 int M, int Nc, int Kd, int lda, int ldb, int ldc,
                 const float* __restrict__ rowmax, const float* __restrict__ rowinv)
{
  __shared__ __bf16 As [64][34];    // [m][k]  (stride 34: even, 4B-aligned pairs)
  __shared__ __bf16 BsT[128][34];   // [n][k]  transposed tile

  const int tid  = threadIdx.x;
  const int wave = tid >> 5;
  const int lane = tid & 31;
  const int rowBase = blockIdx.y * 64;
  const int colBase = blockIdx.x * 128;
  const int wm = wave >> 2;      // 0..1  (two 32-row groups)
  const int wn = wave & 3;       // 0..3  (four 32-col groups)
  const int l  = lane & 15;
  const int hi = lane >> 4;

  v8f acc[2][2];
  #pragma unroll
  for (int a = 0; a < 2; ++a)
    #pragma unroll
    for (int b = 0; b < 2; ++b)
      #pragma unroll
      for (int r = 0; r < 8; ++r) acc[a][b][r] = 0.0f;

  for (int k0 = 0; k0 < Kd; k0 += 32) {
    // ---- stage A tile 64x32 (f32 -> bf16) ----
    #pragma unroll
    for (int r = 0; r < 8; ++r) {
      int idx = tid + r * 256;
      int ar = idx >> 5, ac = idx & 31;
      int gr = rowBase + ar, gc = k0 + ac;
      float v = 0.0f;
      if (ALIGNED || (gr < M && gc < Kd)) {
        v = A[(long long)gr * lda + gc];
        if (FLAGS & 4)  v = __expf(-v - rowmax[gr]) * rowinv[gr];
        if (FLAGS & 16) v = -v;
      }
      As[ar][ac] = (__bf16)v;
    }
    // ---- stage B tile 32x128 into transposed LDS [n][k] ----
    #pragma unroll
    for (int r = 0; r < 16; ++r) {
      int idx = tid + r * 256;
      int bk = idx >> 7, bn = idx & 127;     // k-row, n-col of the tile
      int gk = k0 + bk, gn = colBase + bn;
      float v = 0.0f;
      if (ALIGNED || (gk < Kd && gn < Nc)) {
        if (FLAGS & 8) v = B[(long long)gn * ldb + gk];
        else           v = B[(long long)gk * ldb + gn];
      }
      BsT[bn][bk] = (__bf16)v;
    }
    __syncthreads();

    // ---- build fragments (32-bit LDS loads only) ----
    v16bf afrag[2], bfrag[2];
    #pragma unroll
    for (int mt = 0; mt < 2; ++mt) {
      int mrow = wm * 32 + mt * 16 + l;
      FragU fu;
      #pragma unroll
      for (int i = 0; i < 8; ++i) {
        // A 16x32 bf16: VGPR i, lanes0-15: K = i<4 ? 2i : 16+2(i-4); lanes16-31: +8
        int kb = (i < 4) ? (hi * 8 + 2 * i) : (16 + hi * 8 + 2 * (i - 4));
        fu.u[i] = *(const unsigned*)&As[mrow][kb];
      }
      afrag[mt] = fu.v;
    }
    #pragma unroll
    for (int nt = 0; nt < 2; ++nt) {
      int ncol = wn * 32 + nt * 16 + l;
      FragU fu;
      #pragma unroll
      for (int i = 0; i < 8; ++i) {
        // B 32x16 bf16: VGPR i, lanes0-15: K = i<4 ? 2i : 16+2(i-4); lanes16-31: +8
        int kk2 = (i < 4) ? (hi * 8 + 2 * i) : (16 + hi * 8 + 2 * (i - 4));
        fu.u[i] = *(const unsigned*)&BsT[ncol][kk2];
      }
      bfrag[nt] = fu.v;
    }
    #pragma unroll
    for (int mt = 0; mt < 2; ++mt)
      #pragma unroll
      for (int nt = 0; nt < 2; ++nt)
        acc[mt][nt] = __builtin_amdgcn_wmma_f32_16x16x32_bf16(
            false, afrag[mt], false, bfrag[nt], (short)0, acc[mt][nt],
            false, false);
    __syncthreads();
  }

  // ---- epilogue: C/D f32 layout, VGPR r -> row r (lanes 0-15) / r+8 (16-31) ----
  #pragma unroll
  for (int mt = 0; mt < 2; ++mt) {
    #pragma unroll
    for (int nt = 0; nt < 2; ++nt) {
      #pragma unroll
      for (int r = 0; r < 8; ++r) {
        int row = rowBase + wm * 32 + mt * 16 + r + (hi ? 8 : 0);
        int col = colBase + wn * 32 + nt * 16 + l;
        if (ALIGNED || (row < M && col < Nc)) {
          float v = acc[mt][nt][r];
          if (bias) v += bias[col];
          if (FLAGS & 1) v = fmaxf(v, 0.0f);
          if (FLAGS & 2) v = -v;
          C[(long long)row * ldc + col] = v;
        }
      }
    }
  }
}

// =====================================================================
//  elementwise / graph kernels
// =====================================================================
__global__ void k_zero(float* __restrict__ p, long long n) {
  for (long long i = blockIdx.x * 256LL + threadIdx.x; i < n;
       i += (long long)gridDim.x * 256)
    p[i] = 0.0f;
}

__global__ void k_copy(float* __restrict__ d, const float* __restrict__ s, long long n) {
  for (long long i = blockIdx.x * 256LL + threadIdx.x; i < n;
       i += (long long)gridDim.x * 256)
    d[i] = s[i];
}

__global__ void k_init_scalars(float* scal) {
  if (threadIdx.x == 0 && blockIdx.x == 0) {
    scal[0] = 0.0f; scal[1] = 0.0f; scal[2] = 0.0f;
    ((unsigned*)scal)[3] = 0xFFFFFFFFu;   // ordered-min init
    ((unsigned*)scal)[4] = 0u;            // ordered-max init
  }
}

// SRC=[src,src,dst,dst]  DST=[dst,src,src,dst]
__global__ void k_degrees(const int* __restrict__ src, const int* __restrict__ dst,
                          float* __restrict__ dOut, float* __restrict__ dIn) {
  int i = blockIdx.x * 256 + threadIdx.x;
  if (i >= 4 * EE) return;
  int seg = i >> 16;            // EE = 65536
  int j = i & (EE - 1);
  int sv = src[j], dv = dst[j];
  int s = (seg < 2) ? sv : dv;
  int d = (seg == 0 || seg == 3) ? dv : sv;
  atomicAdd(&dOut[s], 1.0f);
  atomicAdd(&dIn[d], 1.0f);
}

__global__ void k_rsq(float* __restrict__ x, int n) {
  int i = blockIdx.x * 256 + threadIdx.x;
  if (i < n) x[i] = rsqrtf(fmaxf(x[i], 1.0f));
}

__global__ void k_scale_rows(const float* __restrict__ h, const float* __restrict__ rsq,
                             float* __restrict__ o, int fshift, long long total) {
  for (long long i = blockIdx.x * 256LL + threadIdx.x; i < total;
       i += (long long)gridDim.x * 256)
    o[i] = h[i] * rsq[(int)(i >> fshift)];
}

__global__ void k_scatter(const float* __restrict__ hs, const int* __restrict__ src,
                          const int* __restrict__ dst, float* __restrict__ msg,
                          int fshift) {
  long long total = (long long)(4 * EE) << fshift;
  int fmask = (1 << fshift) - 1;
  for (long long i = blockIdx.x * 256LL + threadIdx.x; i < total;
       i += (long long)gridDim.x * 256) {
    int e = (int)(i >> fshift);
    int f = (int)(i & fmask);
    int seg = e >> 16, j = e & (EE - 1);
    int sv = src[j], dv = dst[j];
    int s = (seg < 2) ? sv : dv;
    int d = (seg == 0 || seg == 3) ? dv : sv;
    atomicAdd(&msg[((long long)d << fshift) + f], hs[((long long)s << fshift) + f]);
  }
}

__global__ void k_scale_in(float* __restrict__ msg, const float* __restrict__ rsq,
                           int fshift, long long total) {
  for (long long i = blockIdx.x * 256LL + threadIdx.x; i < total;
       i += (long long)gridDim.x * 256)
    msg[i] *= rsq[(int)(i >> fshift)];
}

// ---------------- BatchNorm (over rows, 256 cols) ----------------
__global__ __launch_bounds__(256)
void k_bn_partial(const float* __restrict__ x, float* __restrict__ bsum,
                  float* __restrict__ bsq, int rows) {
  int c = threadIdx.x;            // one column per thread, coalesced across block
  float s = 0.0f, s2 = 0.0f;
  for (int r = blockIdx.x; r < rows; r += gridDim.x) {
    float v = x[(long long)r * 256 + c];
    s += v; s2 += v * v;
  }
  atomicAdd(&bsum[c], s);
  atomicAdd(&bsq[c], s2);
}

__global__ void k_bn_apply(float* __restrict__ x, const float* __restrict__ bsum,
                           const float* __restrict__ bsq, const float* __restrict__ g,
                           const float* __restrict__ be, int rows) {
  long long total = (long long)rows * 256;
  float inv_n = 1.0f / (float)rows;
  for (long long i = blockIdx.x * 256LL + threadIdx.x; i < total;
       i += (long long)gridDim.x * 256) {
    int c = (int)(i & 255);
    float mu  = bsum[c] * inv_n;
    float var = bsq[c] * inv_n - mu * mu;
    x[i] = (x[i] - mu) * rsqrtf(var + 1e-5f) * g[c] + be[c];
  }
}

// ---------------- softmax stats over dist = -logits ----------------
__global__ __launch_bounds__(256)
void k_rowstats(const float* __restrict__ dist, float* __restrict__ rowmax,
                float* __restrict__ rowinv, float* __restrict__ commit) {
  __shared__ float red[256];
  int row = blockIdx.x, t = threadIdx.x;
  const float* d = dist + (long long)row * KKK;
  float mx = -3.4e38f;
  for (int k = t; k < KKK; k += 256) mx = fmaxf(mx, -d[k]);
  red[t] = mx; __syncthreads();
  for (int s = 128; s > 0; s >>= 1) { if (t < s) red[t] = fmaxf(red[t], red[t + s]); __syncthreads(); }
  mx = red[0]; __syncthreads();
  float se = 0.0f;
  for (int k = t; k < KKK; k += 256) se += __expf(-d[k] - mx);
  red[t] = se; __syncthreads();
  for (int s = 128; s > 0; s >>= 1) { if (t < s) red[t] += red[t + s]; __syncthreads(); }
  float inv = 1.0f / red[0]; __syncthreads();
  float ent = 0.0f;
  for (int k = t; k < KKK; k += 256) {
    float p = __expf(-d[k] - mx) * inv;
    ent -= p * __logf(p + 1e-10f);
  }
  red[t] = ent; __syncthreads();
  for (int s = 128; s > 0; s >>= 1) { if (t < s) red[t] += red[t + s]; __syncthreads(); }
  if (t == 0) { rowmax[row] = mx; rowinv[row] = inv; atomicAdd(commit, red[0]); }
}

// colsum partial: grid.y = row chunks of 256 rows; atomicAdd partials into colsum
__global__ void k_colsum(const float* __restrict__ dist, const float* __restrict__ rowmax,
                         const float* __restrict__ rowinv, float* __restrict__ colsum) {
  int k = blockIdx.x * 256 + threadIdx.x;
  if (k >= KKK) return;
  int r0 = blockIdx.y * 256;
  float s = 0.0f;
  for (int r = r0; r < r0 + 256; ++r)
    s += __expf(-dist[(long long)r * KKK + k] - rowmax[r]) * rowinv[r];
  atomicAdd(&colsum[k], s);
}

__global__ __launch_bounds__(256)
void k_perp(const float* __restrict__ colsum, float* __restrict__ outp) {
  __shared__ float red[256];
  int t = threadIdx.x;
  float s = 0.0f;
  for (int k = t; k < KKK; k += 256) {
    float avg = colsum[k] * (1.0f / (2 * NDSTN));
    s += avg * __logf(avg + 1e-10f);
  }
  red[t] = s; __syncthreads();
  for (int st = 128; st > 0; st >>= 1) { if (t < st) red[t] += red[t + st]; __syncthreads(); }
  if (t == 0) outp[0] = __expf(-red[0]);
}

// ---------------- feature cosine loss ----------------
__global__ __launch_bounds__(128)
void k_cos(const float* __restrict__ feats, const float* __restrict__ qn,
           float* __restrict__ acc) {
  __shared__ float rd0[128], rd1[128], rd2[128];
  int row = blockIdx.x, t = threadIdx.x;
  float f = feats[(long long)row * DINN + t];
  float q = qn[(long long)row * DINN + t];
  rd0[t] = f * q; rd1[t] = f * f; rd2[t] = q * q;
  __syncthreads();
  for (int s = 64; s > 0; s >>= 1) {
    if (t < s) { rd0[t] += rd0[t + s]; rd1[t] += rd1[t + s]; rd2[t] += rd2[t + s]; }
    __syncthreads();
  }
  if (t == 0) {
    float nf = fmaxf(sqrtf(rd1[0]), 1e-12f);
    float nq = fmaxf(sqrtf(rd2[0]), 1e-12f);
    float c = rd0[0] / (nf * nq);
    float e = 1.0f - c;
    atomicAdd(acc, e * e);
  }
}

// ---------------- adjq min/max + edge loss ----------------
__global__ __launch_bounds__(256)
void k_minmax(const float* __restrict__ x, unsigned* __restrict__ uscal, long long n) {
  __shared__ float rmin[256], rmax[256];
  int t = threadIdx.x;
  float mn = 3.4e38f, mx = -3.4e38f;
  for (long long i = blockIdx.x * 256LL + t; i < n; i += (long long)gridDim.x * 256) {
    float v = x[i];
    mn = fminf(mn, v); mx = fmaxf(mx, v);
  }
  rmin[t] = mn; rmax[t] = mx; __syncthreads();
  for (int s = 128; s > 0; s >>= 1) {
    if (t < s) { rmin[t] = fminf(rmin[t], rmin[t + s]); rmax[t] = fmaxf(rmax[t], rmax[t + s]); }
    __syncthreads();
  }
  if (t == 0) { atomicMin(&uscal[3], f2ord(rmin[0])); atomicMax(&uscal[4], f2ord(rmax[0])); }
}

__global__ __launch_bounds__(256)
void k_edge(const float* __restrict__ adj, const float* __restrict__ adjq,
            const unsigned* __restrict__ uscal, float* __restrict__ acc, long long n) {
  __shared__ float red[256];
  float mn = ord2f(uscal[3]);
  float mx = ord2f(uscal[4]);
  float inv = 1.0f / (mx - mn);
  int t = threadIdx.x;
  float s = 0.0f;
  for (long long i = blockIdx.x * 256LL + t; i < n; i += (long long)gridDim.x * 256) {
    float a = (adjq[i] - mn) * inv;
    float d = adj[i] - a;
    s += d * d;
  }
  red[t] = s; __syncthreads();
  for (int st = 128; st > 0; st >>= 1) { if (t < st) red[t] += red[t + st]; __syncthreads(); }
  if (t == 0) atomicAdd(acc, red[0]);
}

__global__ void k_final(const float* __restrict__ scal, float* __restrict__ out_loss) {
  if (threadIdx.x == 0 && blockIdx.x == 0) {
    float commit = 0.25f * scal[0] / (float)(2 * NDSTN);
    float feat   = 10.0f * scal[1] / (float)NDSTN;
    float edge   = sqrtf(scal[2] / ((float)NDSTN * (float)NDSTN));
    out_loss[0] = feat + edge + commit;
  }
}

// =====================================================================
//  host side
// =====================================================================
static inline unsigned gs_blocks(long long n) {
  long long b = (n + 255) / 256;
  if (b > 65536) b = 65536;
  if (b < 1) b = 1;
  return (unsigned)b;
}

static inline void gemm(hipStream_t st, const float* A, const float* B, const float* bias,
                        float* C, int M, int Nc, int Kd, int lda, int ldb, int ldc,
                        const float* rmax, const float* rinv, int flags) {
  dim3 g((unsigned)((Nc + 127) / 128), (unsigned)((M + 63) / 64));
  bool al = (M % 64 == 0) && (Nc % 128 == 0) && (Kd % 32 == 0);
  #define LAUNCH(F)                                                                     \
    do {                                                                                \
      if (al) k_wmma_gemm<F, true ><<<g, 256, 0, st>>>(A, B, bias, C, M, Nc, Kd,        \
                                                       lda, ldb, ldc, rmax, rinv);      \
      else    k_wmma_gemm<F, false><<<g, 256, 0, st>>>(A, B, bias, C, M, Nc, Kd,        \
                                                       lda, ldb, ldc, rmax, rinv);      \
    } while (0)
  switch (flags) {
    case 1:  LAUNCH(1);  break;
    case 2:  LAUNCH(2);  break;
    case 4:  LAUNCH(4);  break;
    case 8:  LAUNCH(8);  break;
    case 16: LAUNCH(16); break;
    default: LAUNCH(0);  break;
  }
  #undef LAUNCH
}

extern "C" void kernel_launch(void* const* d_in, const int* in_sizes, int n_in,
                              void* d_out, int out_size, void* d_ws, size_t ws_size,
                              hipStream_t stream) {
  (void)in_sizes; (void)n_in; (void)out_size; (void)ws_size;
  const float* feats = (const float*)d_in[0];
  const float* adj   = (const float*)d_in[1];
  const int*   src   = (const int*)d_in[2];
  const int*   dst   = (const int*)d_in[3];
  const float* w1 = (const float*)d_in[4];   const float* b1 = (const float*)d_in[5];
  const float* w2 = (const float*)d_in[6];   const float* b2 = (const float*)d_in[7];
  const float* w3 = (const float*)d_in[8];   const float* b3 = (const float*)d_in[9];
  const float* w4 = (const float*)d_in[10];  const float* b4 = (const float*)d_in[11];
  const float* g1 = (const float*)d_in[12];  const float* be1 = (const float*)d_in[13];
  const float* g2 = (const float*)d_in[14];  const float* be2 = (const float*)d_in[15];
  const float* g3 = (const float*)d_in[16];  const float* be3 = (const float*)d_in[17];
  const float* g4 = (const float*)d_in[18];  const float* be4 = (const float*)d_in[19];
  const float* wproj = (const float*)d_in[20]; const float* bproj = (const float*)d_in[21];
  const float* codebook = (const float*)d_in[22];
  const float* wd1 = (const float*)d_in[23]; const float* bd1 = (const float*)d_in[24];
  const float* wd2 = (const float*)d_in[25]; const float* bd2 = (const float*)d_in[26];
  const float* wlin = (const float*)d_in[27]; const float* blin = (const float*)d_in[28];

  float* W   = (float*)d_ws;
  float* out = (float*)d_out;

  float* RSQO = W + WS_RSQO;   float* RSQI = W + WS_RSQI;
  float* HS   = W + WS_HS;     float* MSG  = W + WS_MSG;
  float* HA   = W + WS_HA;     float* HB   = W + WS_HB;
  float* HCAT = W + WS_HCAT;
  float* RMAX = W + WS_ROWMAX; float* RINV = W + WS_ROWINV;
  float* CSUM = W + WS_COLSUM;
  float* QN   = W + WS_QN;     float* QE   = W + WS_QE;
  float* ADJQ = W + WS_ADJQ;
  float* BNS  = W + WS_BNS;    float* BNSQ = W + WS_BNSQ;
  float* SCAL = W + WS_SCAL;

  // ---- degrees (shared by all gconvs) ----
  k_zero<<<gs_blocks(2 * NN), 256, 0, stream>>>(RSQO, 2 * NN);   // RSQO+RSQI contiguous
  k_init_scalars<<<1, 1, 0, stream>>>(SCAL);
  k_degrees<<<(4 * EE) / 256, 256, 0, stream>>>(src, dst, RSQO, RSQI);
  k_rsq<<<NN / 256, 256, 0, stream>>>(RSQO, NN);
  k_rsq<<<NN / 256, 256, 0, stream>>>(RSQI, NN);

  // ---- gconv helper ----
  auto gconv = [&](const float* hin, int F, int fshift, const float* Wt,
                   const float* bt, float* obuf) {
    long long tot = (long long)NN * F;
    k_scale_rows<<<gs_blocks(tot), 256, 0, stream>>>(hin, RSQO, HS, fshift, tot);
    k_zero<<<gs_blocks(tot), 256, 0, stream>>>(MSG, tot);
    k_scatter<<<gs_blocks((long long)(4 * EE) << fshift), 256, 0, stream>>>(HS, src, dst, MSG, fshift);
    k_scale_in<<<gs_blocks(tot), 256, 0, stream>>>(MSG, RSQI, fshift, tot);
    gemm(stream, MSG, Wt, bt, obuf, NN, DHH, F, F, DHH, DHH, nullptr, nullptr, /*relu*/1);
  };
  auto bn = [&](float* buf, const float* g, const float* be) {
    k_zero<<<2, 256, 0, stream>>>(BNS, 2 * DHH);   // BNS+BNSQ contiguous
    k_bn_partial<<<64, 256, 0, stream>>>(buf, BNS, BNSQ, NN);
    k_bn_apply<<<gs_blocks((long long)NN * DHH), 256, 0, stream>>>(buf, BNS, BNSQ, g, be, NN);
  };

  // node branch
  gconv(feats, DINN, 7, w1, b1, HA);  bn(HA, g1, be1);
  gconv(HA, DHH, 8, w2, b2, HB);      bn(HB, g2, be2);
  k_copy<<<gs_blocks((long long)NDSTN * DHH), 256, 0, stream>>>(HCAT, HB, (long long)NDSTN * DHH);
  // link branch
  gconv(feats, DINN, 7, w3, b3, HA);  bn(HA, g3, be3);
  gconv(HA, DHH, 8, w4, b4, HB);      bn(HB, g4, be4);
  k_copy<<<gs_blocks((long long)NDSTN * DHH), 256, 0, stream>>>(HCAT + (long long)NDSTN * DHH, HB,
                                                                (long long)NDSTN * DHH);

  // dist = -(h @ wproj + bproj)   (8192 x 4096), written straight into d_out
  float* DIST = out + O_DIST;
  gemm(stream, HCAT, wproj, bproj, DIST, 2 * NDSTN, KKK, DHH, DHH, KKK, KKK,
       nullptr, nullptr, /*negOut*/2);

  // softmax row stats + commit entropy, column sums, perplexity
  k_rowstats<<<2 * NDSTN, 256, 0, stream>>>(DIST, RMAX, RINV, &SCAL[0]);
  k_zero<<<gs_blocks(KKK), 256, 0, stream>>>(CSUM, KKK);
  {
    dim3 g(KKK / 256, (2 * NDSTN) / 256);   // (16, 32)
    k_colsum<<<g, 256, 0, stream>>>(DIST, RMAX, RINV, CSUM);
  }
  k_perp<<<1, 256, 0, stream>>>(CSUM, out + O_PERP);

  // quantized = softmax(-dist) @ codebook   (8192 x 256) -> d_out
  float* QUANT = out + O_QUANT;
  gemm(stream, DIST, codebook, nullptr, QUANT, 2 * NDSTN, DHH, KKK, KKK, DHH, DHH,
       RMAX, RINV, /*softmaxA*/4);

  // qn = quantized[:4096] @ wd1 + bd1 ; feature cosine loss
  gemm(stream, QUANT, wd1, bd1, QN, NDSTN, DINN, DHH, DHH, DINN, DINN, nullptr, nullptr, 0);
  k_cos<<<NDSTN, 128, 0, stream>>>(feats, QN, &SCAL[1]);

  // qe = quantized[4096:] @ wd2 + bd2 ; adjq = qe @ qe^T ; edge loss
  gemm(stream, QUANT + (long long)NDSTN * DHH, wd2, bd2, QE, NDSTN, DHH, DHH,
       DHH, DHH, DHH, nullptr, nullptr, 0);
  gemm(stream, QE, QE, nullptr, ADJQ, NDSTN, NDSTN, DHH, DHH, DHH, NDSTN,
       nullptr, nullptr, /*transB*/8);
  long long nadj = (long long)NDSTN * NDSTN;
  k_minmax<<<2048, 256, 0, stream>>>(ADJQ, (unsigned*)SCAL, nadj);
  k_edge<<<2048, 256, 0, stream>>>(adj, ADJQ, (const unsigned*)SCAL, &SCAL[2], nadj);

  // hout = (logits @ wlin + blin)[:4096]  with logits = -dist
  gemm(stream, DIST, wlin, blin, out + O_HOUT, NDSTN, DOUTN, KKK, KKK, DOUTN, DOUTN,
       nullptr, nullptr, /*negA*/16);

  // loss scalar
  k_final<<<1, 1, 0, stream>>>(SCAL, out + O_LOSS);
}
